// Head_18330920419536
// MI455X (gfx1250) — matrix operand order repaired and checked by reading, not compiled, not run on hardware
//
#include <hip/hip_runtime.h>
#include <math.h>

// ---------------------------------------------------------------------------
// Fused causal attention head for MI455X (gfx1250, wave32, WMMA).
//   Pass 0: repack Wq|Wk|Wv (f32) into f16 B-fragment order (one-shot, tiny).
//   Pass 1: fused QKV projection — x read ONCE; 12 WMMAs per 32-wide K chunk.
//           q stored pre-scaled by 1/sqrt(C); v stored transposed [b][h][t].
//   Pass 2: flash attention, 16 query rows/wave, 32-key tiles, online softmax.
// All matmul FLOPs via v_wmma_f32_16x16x32_f16; all hot-loop fragment loads
// are aligned 16/32-byte vector loads (global_load_b128).
// ---------------------------------------------------------------------------

typedef __attribute__((ext_vector_type(16))) _Float16 v16h;
typedef __attribute__((ext_vector_type(8)))  _Float16 v8h;
typedef __attribute__((ext_vector_type(8)))  float    v8f;

constexpr int B_  = 8;
constexpr int T_  = 2048;
constexpr int C_  = 1024;
constexpr int H_  = 64;
constexpr int BT_ = B_ * T_;             // 16384 rows
constexpr float SCALE = 0.03125f;        // 1/sqrt(1024)

// ISA 7.12.2: 16-bit A-matrix 16x32 fragment element i (0..15) -> K index.
__device__ __forceinline__ int mapA(int i, int half) {
  return (i < 8) ? (i + 8 * half) : (i + 8 + 8 * half);
}

// ---------------------------------------------------------------------------
// Pass 0: repack weights into B-fragment-major f16: Wp[w][kc][nt][lane][i].
// B-frag map (32x16 16-bit): lanes 0-15 hold K=0..7,16..23; lanes 16-31 K=8..15,24..31.
// ---------------------------------------------------------------------------
__global__ __launch_bounds__(256)
void repack_w_kernel(const float* __restrict__ Wq,
                     const float* __restrict__ Wk,
                     const float* __restrict__ Wv,
                     _Float16* __restrict__ Wp) {
  const int idx = blockIdx.x * 256 + threadIdx.x;
  if (idx >= (3 << 16)) return;          // 3 * 32kc * 4nt * 32lane * 16
  const int i    = idx & 15;
  const int lane = (idx >> 4) & 31;
  const int nt   = (idx >> 9) & 3;
  const int kc   = (idx >> 11) & 31;
  const int w    = idx >> 16;
  const float* __restrict__ W = (w == 0) ? Wq : (w == 1 ? Wk : Wv);
  const int half = lane >> 4;
  const int lm   = lane & 15;
  const int k    = mapA(i, half);        // B-frag K map (same as A table order)
  Wp[idx] = (_Float16)W[(size_t)(kc * 32 + k) * H_ + nt * 16 + lm];
}

// ---------------------------------------------------------------------------
// Pass 1: fused QKV projection. grid = BT/16, block = 32 (one wave).
// One x A-fragment per K chunk feeds 12 WMMAs (q,k,v x 4 col-tiles).
// ---------------------------------------------------------------------------
__global__ __launch_bounds__(32)
void qkv_proj_kernel(const float* __restrict__ x,
                     const v16h* __restrict__ Wp,   // fragment-major weights
                     _Float16* __restrict__ qo,     // [b*t][h], pre-scaled
                     _Float16* __restrict__ ko,     // [b*t][h]
                     _Float16* __restrict__ vT) {   // [b][h][t] (transposed)
  const int tile = blockIdx.x;
  const int lane = threadIdx.x & 31;
  const int half = lane >> 4;
  const int lm   = lane & 15;
  const int row0 = tile * 16;

  v8f acc[3][4] = {};                    // [q|k|v][4 col tiles], 16x16 f32

  for (int kc = 0; kc < C_ / 32; ++kc) {
    // A fragment: 2x contiguous 8-float runs per lane -> 4x b128 + cvt_pk
    const float* xp = x + (size_t)(row0 + lm) * C_ + kc * 32 + 8 * half;
    const float4 f0 = *(const float4*)(xp);
    const float4 f1 = *(const float4*)(xp + 4);
    const float4 f2 = *(const float4*)(xp + 16);
    const float4 f3 = *(const float4*)(xp + 20);
    v16h a;
    a[0]=(_Float16)f0.x;  a[1]=(_Float16)f0.y;  a[2]=(_Float16)f0.z;  a[3]=(_Float16)f0.w;
    a[4]=(_Float16)f1.x;  a[5]=(_Float16)f1.y;  a[6]=(_Float16)f1.z;  a[7]=(_Float16)f1.w;
    a[8]=(_Float16)f2.x;  a[9]=(_Float16)f2.y;  a[10]=(_Float16)f2.z; a[11]=(_Float16)f2.w;
    a[12]=(_Float16)f3.x; a[13]=(_Float16)f3.y; a[14]=(_Float16)f3.z; a[15]=(_Float16)f3.w;

#pragma unroll
    for (int w = 0; w < 3; ++w) {
#pragma unroll
      for (int nt = 0; nt < 4; ++nt) {
        const v16h b = Wp[((size_t)((w * 32 + kc) * 4 + nt)) * 32 + lane];
        acc[w][nt] = __builtin_amdgcn_wmma_f32_16x16x32_f16(
            false, a, false, b, (short)0, acc[w][nt], false, false);
      }
    }
  }

  // C/D layout: row = r + 8*half, col = nt*16 + lane%16
#pragma unroll
  for (int nt = 0; nt < 4; ++nt) {
#pragma unroll
    for (int r = 0; r < 8; ++r) {
      const int row = row0 + r + 8 * half;
      const int col = nt * 16 + lm;
      qo[(size_t)row * H_ + col] = (_Float16)(acc[0][nt][r] * SCALE);
      ko[(size_t)row * H_ + col] = (_Float16)acc[1][nt][r];
      const int bb = row >> 11;          // row / T_
      const int t  = row & (T_ - 1);     // row % T_
      vT[((size_t)bb * H_ + col) * T_ + t] = (_Float16)acc[2][nt][r];
    }
  }
}

// ---------------------------------------------------------------------------
// Pass 2: flash attention. grid = (T/16, B), block = 32 (one wave).
// ---------------------------------------------------------------------------
__global__ __launch_bounds__(32)
void attn_kernel(const _Float16* __restrict__ q,
                 const _Float16* __restrict__ k,
                 const _Float16* __restrict__ vT,
                 float* __restrict__ out) {
  __shared__ __align__(32) _Float16 lds_p[16 * 32];   // P bounce buffer (1 KB)

  const int b     = blockIdx.y;
  const int qt    = blockIdx.x;
  const int lane  = threadIdx.x & 31;
  const int half  = lane >> 4;
  const int lm    = lane & 15;
  const int qbase = qt * 16;

  const _Float16* __restrict__ qb = q  + (size_t)b * T_ * H_;
  const _Float16* __restrict__ kb = k  + (size_t)b * T_ * H_;
  const _Float16* __restrict__ vb = vT + (size_t)b * H_ * T_;

  // Q A-frags (already scaled by 1/sqrt(C)): two v8h runs -> v16h each
  v16h aq[2];
#pragma unroll
  for (int c = 0; c < 2; ++c) {
    const _Float16* qp = qb + (size_t)(qbase + lm) * H_ + c * 32 + 16 * half;
    const v8h lo = *(const v8h*)(qp);
    const v8h hi = *(const v8h*)(qp + 8);
    aq[c] = __builtin_shufflevector(lo, hi, 0,1,2,3,4,5,6,7,8,9,10,11,12,13,14,15);
  }

  float mstat[8], lstat[8];
#pragma unroll
  for (int r = 0; r < 8; ++r) { mstat[r] = -1e30f; lstat[r] = 0.0f; }
  v8f o[4] = {};

  const int ntiles = (qbase + 16 + 31) / 32;   // covers keys 0 .. qbase+15
  for (int jt = 0; jt < ntiles; ++jt) {
    const int jbase = jt * 32;

    if (jt + 1 < ntiles) {               // prefetch next K/V tiles (L2 hot)
      __builtin_prefetch(kb + (size_t)(jbase + 32) * H_, 0, 1);
      __builtin_prefetch(vb + (size_t)lm * T_ + jbase + 32, 0, 1);
    }

    // ---- S = Q*K^T : two 16x16 tiles, K-dim = 64 (2 chunks) -> 4 WMMA ----
    v8f s[2] = {};
#pragma unroll
    for (int nt = 0; nt < 2; ++nt) {
#pragma unroll
      for (int c = 0; c < 2; ++c) {
        const v16h bk = *(const v16h*)(kb + (size_t)(jbase + nt * 16 + lm) * H_
                                          + c * 32 + 16 * half);
        s[nt] = __builtin_amdgcn_wmma_f32_16x16x32_f16(
            false, aq[c], false, bk, (short)0, s[nt], false, false);
      }
    }

    // ---- causal mask (scale already folded into q) ----
#pragma unroll
    for (int nt = 0; nt < 2; ++nt) {
#pragma unroll
      for (int r = 0; r < 8; ++r) {
        const int key  = jbase + nt * 16 + lm;
        const int qrow = qbase + r + 8 * half;
        s[nt][r] = (key > qrow) ? -1e30f : s[nt][r];
      }
    }

    // ---- online softmax update ----
#pragma unroll
    for (int r = 0; r < 8; ++r) {
      float rm = fmaxf(s[0][r], s[1][r]);
#pragma unroll
      for (int m = 1; m < 16; m <<= 1)
        rm = fmaxf(rm, __shfl_xor(rm, m, 16));

      const float mnew  = fmaxf(mstat[r], rm);
      const float alpha = __expf(mstat[r] - mnew);
      const float p0 = __expf(s[0][r] - mnew);
      const float p1 = __expf(s[1][r] - mnew);

      float rs = p0 + p1;
#pragma unroll
      for (int m = 1; m < 16; m <<= 1)
        rs += __shfl_xor(rs, m, 16);

      lstat[r] = lstat[r] * alpha + rs;
      mstat[r] = mnew;
#pragma unroll
      for (int nt = 0; nt < 4; ++nt) o[nt][r] *= alpha;

      const int row = r + 8 * half;
      lds_p[row * 32 +      lm] = (_Float16)p0;
      lds_p[row * 32 + 16 + lm] = (_Float16)p1;
    }
    __syncthreads();   // single-wave WG: LDS write->transposed-read fence

    // ---- O += P*V : A = P from LDS (2x ds b128), B = V^T rows (b128) ----
    const _Float16* pp = lds_p + lm * 32 + 16 * half;
    const v8h plo = *(const v8h*)(pp);
    const v8h phi = *(const v8h*)(pp + 8);
    const v16h ap = __builtin_shufflevector(plo, phi,
                        0,1,2,3,4,5,6,7,8,9,10,11,12,13,14,15);

#pragma unroll
    for (int nt = 0; nt < 4; ++nt) {
      const v16h bv = *(const v16h*)(vb + (size_t)(nt * 16 + lm) * T_
                                        + jbase + 16 * half);
      o[nt] = __builtin_amdgcn_wmma_f32_16x16x32_f16(
          false, ap, false, bv, (short)0, o[nt], false, false);
    }
    __syncthreads();   // protect lds_p before next iteration
  }

  // ---- epilogue: normalize by row sums, store f32 ----
#pragma unroll
  for (int nt = 0; nt < 4; ++nt) {
#pragma unroll
    for (int r = 0; r < 8; ++r) {
      const int row = qbase + r + 8 * half;
      const int col = nt * 16 + lm;
      out[((size_t)b * T_ + row) * H_ + col] = o[nt][r] / lstat[r];
    }
  }
}

// ---------------------------------------------------------------------------
extern "C" void kernel_launch(void* const* d_in, const int* in_sizes, int n_in,
                              void* d_out, int out_size, void* d_ws, size_t ws_size,
                              hipStream_t stream) {
  const float* x  = (const float*)d_in[0];
  const float* Wq = (const float*)d_in[1];
  const float* Wk = (const float*)d_in[2];
  const float* Wv = (const float*)d_in[3];

  _Float16* ws   = (_Float16*)d_ws;
  _Float16* qbuf = ws;                              // 2 MB
  _Float16* kbuf = ws + (size_t)BT_ * H_;           // 2 MB
  _Float16* vbuf = ws + (size_t)2 * BT_ * H_;       // 2 MB (transposed)
  _Float16* Wp   = ws + (size_t)3 * BT_ * H_;       // 384 KB fragment-major W

  repack_w_kernel<<<((3 << 16) + 255) / 256, 256, 0, stream>>>(Wq, Wk, Wv, Wp);

  qkv_proj_kernel<<<BT_ / 16, 32, 0, stream>>>(
      x, (const v16h*)Wp, qbuf, kbuf, vbuf);

  attn_kernel<<<dim3(T_ / 16, B_), 32, 0, stream>>>(
      qbuf, kbuf, vbuf, (float*)d_out);
}
